// GCL_4140348473947
// MI455X (gfx1250) — matrix operand order repaired and checked by reading, not compiled, not run on hardware
//
#include <hip/hip_runtime.h>
#include <hip/hip_bf16.h>

// ---------------------------------------------------------------------------
// EGNN GCL layer for MI455X (gfx1250, wave32).
// bf16 WMMA (v_wmma_f32_16x16x32_bf16) with fp32 accumulate.
//  d_out = [ h_out (N*128 f32) | mij (E*128 f32) ]
//  h_out region doubles as the fp32 "agg" scatter buffer (zero -> atomic
//  scatter -> consumed+overwritten in place by the node kernel).
//  Edge kernel register-blocks M: 2 x 16-edge sub-tiles per wave so each
//  B fragment load feeds two WMMAs (halves L0 weight traffic).
// ---------------------------------------------------------------------------

typedef __attribute__((ext_vector_type(16))) __bf16 bf16x16;
typedef __attribute__((ext_vector_type(8)))  __bf16 bf16x8;
typedef __attribute__((ext_vector_type(8)))  float  f32x8;
typedef __attribute__((ext_vector_type(4)))  float  f32x4;

#define D_NF      128
#define LDS_PITCH 136   // 128 + 8 bf16 pad -> 272B row stride, conflict-free b128 reads
#define NORM_INV  0.01f

// ---- A-matrix fragment (16x32 bf16, ISA 7.12.2 layout) --------------------
// Lane (m = lane&15, hf = lane>>4): elements 0..7  = channels hf*8 + 0..7
//                                   elements 8..15 = channels hf*8 + 16..23
__device__ __forceinline__ bf16x16 afrag_from(const __bf16* p) {
  bf16x8 lo = *(const bf16x8*)(p);
  bf16x8 hi = *(const bf16x8*)(p + 16);
  bf16x16 a;
#pragma unroll
  for (int i = 0; i < 8; ++i) { a[i] = lo[i]; a[i + 8] = hi[i]; }
  return a;
}

__device__ __forceinline__ float silu(float x) {
  return x / (1.0f + __expf(-x));
}

// ---- one-time prep: h -> bf16, zero agg ----------------------------------
__global__ void init_kernel(const float* __restrict__ h, __bf16* __restrict__ hbf,
                            float* __restrict__ agg, int total) {
  int gid = blockIdx.x * blockDim.x + threadIdx.x;
  if (gid < total) {
    hbf[gid] = (__bf16)h[gid];
    agg[gid] = 0.0f;
  }
}

// ---- one-time prep: weights -> fragment-major bf16 B operands -------------
// B (32x16 bf16, dense): lane (n = lane&15, hb = lane>>4) holds K = hb*8 + i (i<8),
// 16 + hb*8 + (i-8) (i>=8) of column n. Fragment buffer layout:
//   frag[((kk*8 + n)*32 + lane)*16 + i]  -> each lane loads 32 contiguous B.
__global__ void pack_weights(const float* __restrict__ We1, const float* __restrict__ We2,
                             const float* __restrict__ Wn1, const float* __restrict__ Wn2,
                             __bf16* __restrict__ fragBase) {
  int gid = blockIdx.x * blockDim.x + threadIdx.x;
  if (gid >= 98304) return;                 // (64+32+64+32) frags * 512 elems
  const float* src;
  int local = gid;
  if      (gid < 32768) { src = We1; }
  else if (gid < 49152) { src = We2; local = gid - 32768; }
  else if (gid < 81920) { src = Wn1; local = gid - 49152; }
  else                  { src = Wn2; local = gid - 81920; }
  int fi   = local >> 9;                    // fragment = kk*8 + n
  int rem  = local & 511;
  int lane = rem >> 4;
  int i    = rem & 15;
  int kk   = fi >> 3;
  int n    = fi & 7;
  int K = kk * 32 + (lane >> 4) * 8 + (i & 7) + (i >> 3) * 16;
  int N = n * 16 + (lane & 15);
  fragBase[gid] = (__bf16)src[K * D_NF + N];
}

// ---- edge model: mij = silu(silu([h[row]|h[col]] @ We1 + be1) @ We2 + be2)
// plus scatter: agg[row] += mij / NORM.  One wave per 32 edges (2 M-tiles).
__global__ void __launch_bounds__(32)
edge_kernel(const int* __restrict__ ei, const __bf16* __restrict__ hbf,
            const __bf16* __restrict__ we1f, const __bf16* __restrict__ we2f,
            const float* __restrict__ be1, const float* __restrict__ be2,
            float* agg, float* __restrict__ mij, int E) {
  __shared__ alignas(16) __bf16 xs[32][LDS_PITCH];
  const int lane = threadIdx.x & 31;
  const int m    = lane & 15;
  const int hf   = lane >> 4;
  const int tile = blockIdx.x * 32;          // two 16-edge sub-tiles

  const int* rowIdx = ei;          // edge_index[0]
  const int* colIdx = ei + E;      // edge_index[1]
  int r[2], c[2];
  r[0] = rowIdx[tile + m];       c[0] = colIdx[tile + m];
  r[1] = rowIdx[tile + 16 + m];  c[1] = colIdx[tile + 16 + m];

  // ---- GEMM1: [2 x 16 x 256] @ We1 -> [2 x 16 x 128], K = 8 steps --------
  f32x8 acc[2][8] = {};
#pragma unroll
  for (int kk = 0; kk < 8; ++kk) {
    const int n0 = (kk < 4) ? r[0] : c[0];           // channels <128: src row
    const int n1 = (kk < 4) ? r[1] : c[1];
    const size_t koff = (size_t)((kk & 3) * 32 + hf * 8);
    bf16x16 a0 = afrag_from(hbf + (size_t)n0 * D_NF + koff);
    bf16x16 a1 = afrag_from(hbf + (size_t)n1 * D_NF + koff);
#pragma unroll
    for (int n = 0; n < 8; ++n) {
      bf16x16 b = *(const bf16x16*)(we1f + (size_t)(kk * 8 + n) * 512 + lane * 16);
      acc[0][n] = __builtin_amdgcn_wmma_f32_16x16x32_bf16(
          false, a0, false, b, (short)0, acc[0][n], false, false);
      acc[1][n] = __builtin_amdgcn_wmma_f32_16x16x32_bf16(
          false, a1, false, b, (short)0, acc[1][n], false, false);
    }
  }

  // bias + SiLU, transpose D layout (col-per-lane) -> A layout (row-per-lane)
#pragma unroll
  for (int t = 0; t < 2; ++t) {
#pragma unroll
    for (int n = 0; n < 8; ++n) {
      float bias = be1[n * 16 + m];
#pragma unroll
      for (int p = 0; p < 8; ++p) {
        float x = acc[t][n][p] + bias;       // (M = hf*8+p, N = n*16+m)
        xs[t * 16 + hf * 8 + p][n * 16 + m] = (__bf16)silu(x);
      }
    }
  }
  __syncthreads();

  // ---- GEMM2: [2 x 16 x 128] @ We2 -> [2 x 16 x 128], K = 4 steps --------
  f32x8 acc2[2][8] = {};
#pragma unroll
  for (int kk = 0; kk < 4; ++kk) {
    bf16x16 a0 = afrag_from(&xs[m][kk * 32 + hf * 8]);
    bf16x16 a1 = afrag_from(&xs[16 + m][kk * 32 + hf * 8]);
#pragma unroll
    for (int n = 0; n < 8; ++n) {
      bf16x16 b = *(const bf16x16*)(we2f + (size_t)(kk * 8 + n) * 512 + lane * 16);
      acc2[0][n] = __builtin_amdgcn_wmma_f32_16x16x32_bf16(
          false, a0, false, b, (short)0, acc2[0][n], false, false);
      acc2[1][n] = __builtin_amdgcn_wmma_f32_16x16x32_bf16(
          false, a1, false, b, (short)0, acc2[1][n], false, false);
    }
  }

  // bias + SiLU -> store mij, atomic scatter mij/NORM into agg
#pragma unroll
  for (int t = 0; t < 2; ++t) {
    int destRow[8];
#pragma unroll
    for (int p = 0; p < 8; ++p) destRow[p] = rowIdx[tile + t * 16 + hf * 8 + p];
#pragma unroll
    for (int n = 0; n < 8; ++n) {
      const int col = n * 16 + m;
      const float bias = be2[col];
#pragma unroll
      for (int p = 0; p < 8; ++p) {
        const int e = tile + t * 16 + hf * 8 + p;
        float v = silu(acc2[t][n][p] + bias);
        mij[(size_t)e * D_NF + col] = v;
        atomicAdd(&agg[(size_t)destRow[p] * D_NF + col], v * NORM_INV);
      }
    }
  }
}

// ---- node model: h_out = h + silu([h|agg] @ Wn1 + bn1) @ Wn2 + bn2 -------
// houtagg is both agg input (already /NORM) and h_out output (in place).
__global__ void __launch_bounds__(32)
node_kernel(const float* __restrict__ h, const __bf16* __restrict__ hbf,
            float* houtagg,
            const __bf16* __restrict__ wn1f, const __bf16* __restrict__ wn2f,
            const float* __restrict__ bn1, const float* __restrict__ bn2) {
  __shared__ alignas(16) __bf16 xs[16][LDS_PITCH];
  const int lane = threadIdx.x & 31;
  const int m    = lane & 15;
  const int hf   = lane >> 4;
  const int tile = blockIdx.x * 16;
  const int node = tile + m;

  f32x8 acc[8] = {};
#pragma unroll
  for (int kk = 0; kk < 8; ++kk) {
    bf16x16 a;
    if (kk < 4) {                                   // channels 0..127 : h (bf16)
      a = afrag_from(hbf + (size_t)node * D_NF + kk * 32 + hf * 8);
    } else {                                        // channels 128..255 : agg (f32)
      const float* ap = houtagg + (size_t)node * D_NF + (kk - 4) * 32 + hf * 8;
      f32x4 f0 = *(const f32x4*)(ap);
      f32x4 f1 = *(const f32x4*)(ap + 4);
      f32x4 f2 = *(const f32x4*)(ap + 16);
      f32x4 f3 = *(const f32x4*)(ap + 20);
#pragma unroll
      for (int i = 0; i < 4; ++i) {
        a[i]      = (__bf16)f0[i];
        a[i + 4]  = (__bf16)f1[i];
        a[i + 8]  = (__bf16)f2[i];
        a[i + 12] = (__bf16)f3[i];
      }
    }
#pragma unroll
    for (int n = 0; n < 8; ++n) {
      bf16x16 b = *(const bf16x16*)(wn1f + (size_t)(kk * 8 + n) * 512 + lane * 16);
      acc[n] = __builtin_amdgcn_wmma_f32_16x16x32_bf16(
          false, a, false, b, (short)0, acc[n], false, false);
    }
  }

#pragma unroll
  for (int n = 0; n < 8; ++n) {
    float bias = bn1[n * 16 + m];
#pragma unroll
    for (int p = 0; p < 8; ++p) {
      float x = acc[n][p] + bias;
      xs[hf * 8 + p][n * 16 + m] = (__bf16)silu(x);
    }
  }
  __syncthreads();

  f32x8 acc2[8] = {};
#pragma unroll
  for (int kk = 0; kk < 4; ++kk) {
    bf16x16 a = afrag_from(&xs[m][kk * 32 + hf * 8]);
#pragma unroll
    for (int n = 0; n < 8; ++n) {
      bf16x16 b = *(const bf16x16*)(wn2f + (size_t)(kk * 8 + n) * 512 + lane * 16);
      acc2[n] = __builtin_amdgcn_wmma_f32_16x16x32_bf16(
          false, a, false, b, (short)0, acc2[n], false, false);
    }
  }

  // layer 2 has no activation; add bias + residual, overwrite agg with h_out
#pragma unroll
  for (int n = 0; n < 8; ++n) {
    const int col = n * 16 + m;
    const float bias = bn2[col];
#pragma unroll
    for (int p = 0; p < 8; ++p) {
      const int nd = tile + hf * 8 + p;
      float v = acc2[n][p] + bias;
      houtagg[(size_t)nd * D_NF + col] = h[(size_t)nd * D_NF + col] + v;
    }
  }
}

// ---------------------------------------------------------------------------
extern "C" void kernel_launch(void* const* d_in, const int* in_sizes, int n_in,
                              void* d_out, int out_size, void* d_ws, size_t ws_size,
                              hipStream_t stream) {
  (void)n_in; (void)out_size; (void)ws_size;
  const float* h   = (const float*)d_in[0];
  const int*   ei  = (const int*)  d_in[1];   // [2, E] (row | col)
  const float* We1 = (const float*)d_in[2];
  const float* be1 = (const float*)d_in[3];
  const float* We2 = (const float*)d_in[4];
  const float* be2 = (const float*)d_in[5];
  const float* Wn1 = (const float*)d_in[6];
  const float* bn1 = (const float*)d_in[7];
  const float* Wn2 = (const float*)d_in[8];
  const float* bn2 = (const float*)d_in[9];

  const int N = in_sizes[0] / D_NF;   // 50000
  const int E = in_sizes[1] / 2;      // 800000

  float* out     = (float*)d_out;
  float* agg_out = out;                        // h_out region doubles as agg
  float* mij     = out + (size_t)N * D_NF;     // second tuple element

  // workspace layout (bf16): packed weight fragments, then bf16 h
  __bf16* we1f = (__bf16*)d_ws;                // 64 frags * 512
  __bf16* we2f = we1f + 64 * 512;              // 32 frags * 512
  __bf16* wn1f = we2f + 32 * 512;              // 64 frags * 512
  __bf16* wn2f = wn1f + 64 * 512;              // 32 frags * 512
  __bf16* hbf  = wn2f + 32 * 512;              // N * 128

  const int totalH = N * D_NF;
  init_kernel<<<(totalH + 255) / 256, 256, 0, stream>>>(h, hbf, agg_out, totalH);
  pack_weights<<<(98304 + 255) / 256, 256, 0, stream>>>(We1, We2, Wn1, Wn2, we1f);
  edge_kernel<<<E / 32, 32, 0, stream>>>(ei, hbf, we1f, we2f, be1, be2,
                                         agg_out, mij, E);
  node_kernel<<<N / 16, 32, 0, stream>>>(h, hbf, agg_out, wn1f, wn2f, bn1, bn2);
}
